// Gemma3Attention_38044820308332
// MI455X (gfx1250) — matrix operand order repaired and checked by reading, not compiled, not run on hardware
//
#include <hip/hip_runtime.h>
#include <hip/hip_bf16.h>

// ---------------------------------------------------------------------------
// Gemma3-style attention block for gfx1250 (MI455X).
// bf16 WMMA for all matmuls; TDM (tensor_load_to_lds) for flash-attn staging.
// Shapes (fixed): B=2, T=1024, H=8, KVH=4, D=256, HIDDEN=2560, S=4096,
// WINDOW=1024 == T  ->  attention window == exactly the new K/V block.
// ---------------------------------------------------------------------------

typedef __attribute__((ext_vector_type(16))) __bf16          v16bf;
typedef __attribute__((ext_vector_type(16))) unsigned short  v16u;
typedef __attribute__((ext_vector_type(8)))  float           v8f;
typedef __attribute__((ext_vector_type(4)))  unsigned int    v4u;
typedef __attribute__((ext_vector_type(4)))  int             v4i;
typedef __attribute__((ext_vector_type(8)))  int             v8i_;

#define SCALE   0.0625f        // 256^-0.5
#define B_      2
#define T_      1024
#define NH      8
#define NKV     4
#define HD      256
#define HID     2560
#define SCACHE  4096

__device__ __forceinline__ unsigned short f32_to_bf16(float f) {
    unsigned u = __builtin_bit_cast(unsigned, f);
    unsigned r = u + 0x7FFFu + ((u >> 16) & 1u);   // round-to-nearest-even
    return (unsigned short)(r >> 16);
}
__device__ __forceinline__ unsigned short cvt_bf16(float f)          { return f32_to_bf16(f); }
__device__ __forceinline__ unsigned short cvt_bf16(unsigned short s) { return s; }

// D = A(16x32 bf16) * B(32x16 bf16) + C(16x16 f32); wave32, EXEC all ones.
__device__ __forceinline__ v8f wmma_bf16(v16u a, v16u b, v8f c) {
    return __builtin_amdgcn_wmma_f32_16x16x32_bf16(
        false, __builtin_bit_cast(v16bf, a),
        false, __builtin_bit_cast(v16bf, b),
        (short)0, c, false, false);
}

// ---------------------------------------------------------------------------
// Tensor Data Mover: 2D bf16 tile (tile_d0 elems/row x tile_d1 rows) from
// global -> LDS, with optional LDS pad swizzle. D# layout per ISA 08 §8.3/8.4.
// pad_interval encoded: k -> 2^(k+1) DWORDs; pad_amount encoded: k -> k+1 DWORDs.
// ---------------------------------------------------------------------------
__device__ __forceinline__ void tdm_load_2d_bf16(
    unsigned lds_off, const unsigned short* gptr,
    unsigned tile_d0, unsigned tile_d1,
    unsigned tensor_d0, unsigned tensor_d1, unsigned stride0,
    unsigned pad_interval, unsigned pad_amount) {
    unsigned long long ga = (unsigned long long)(uintptr_t)gptr;
    v4u g0;
    g0.x = 1u;                                          // count=1, user mode
    g0.y = lds_off;                                     // LDS byte address
    g0.z = (unsigned)ga;                                // global_addr[31:0]
    g0.w = (unsigned)((ga >> 32) & 0x1FFFFFFu) | (2u << 30);  // [56:32] | type=2
    unsigned dw0 = (1u << 16)                           // data_size = 1 (2 bytes)
                 | (1u << 20)                           // pad_enable
                 | (pad_interval << 22) | (pad_amount << 25);
    v8i_ g1;
    g1[0] = (int)dw0;
    g1[1] = (int)(tensor_d0 << 16);                     // tensor_dim0[15:0]
    g1[2] = (int)((tensor_d0 >> 16) | (tensor_d1 << 16));
    g1[3] = (int)((tensor_d1 >> 16) | (tile_d0 << 16));
    g1[4] = (int)tile_d1;                               // tile_dim1 | tile_dim2=0
    g1[5] = (int)stride0;                               // tensor_dim0_stride[31:0]
    g1[6] = 0;
    g1[7] = 0;
    v4i z4 = {0, 0, 0, 0};
#if defined(__clang_major__) && (__clang_major__ >= 23)
    v8i_ z8 = {0, 0, 0, 0, 0, 0, 0, 0};
    __builtin_amdgcn_tensor_load_to_lds(g0, g1, z4, z4, z8, 0);
#else
    __builtin_amdgcn_tensor_load_to_lds(g0, g1, z4, z4, 0);
#endif
}

// ---------------------------------------------------------------------------
// Generic tiled GEMM: C[M,N](f32) = A[M,K] * B[K,N], row-major.
// Block: 256 threads = 8 waves; block tile 128x128; wave tile 32x64; K-step 32.
// ---------------------------------------------------------------------------
template <typename TA, typename TB>
__global__ __launch_bounds__(256) void gemm_wmma(
    const TA* __restrict__ A, const TB* __restrict__ B, float* __restrict__ C,
    int M, int N, int K) {
    __shared__ unsigned short As[128][40];
    __shared__ unsigned short Bs[32][136];

    const int tid  = threadIdx.x;
    const int l    = tid & 31;
    const int wave = tid >> 5;
    const int wm   = wave >> 1;
    const int wn   = wave & 1;
    const int m0   = blockIdx.y * 128;
    const int n0   = blockIdx.x * 128;

    v8f acc[2][4] = {};

    const int arow = l & 15;
    const int kh   = (l >> 4) << 3;
    const int bcol = l & 15;
    const int kb   = (l >> 4) << 3;

    for (int k0 = 0; k0 < K; k0 += 32) {
        #pragma unroll
        for (int e = 0; e < 16; ++e) {
            int idx = tid + 256 * e;
            int r = idx >> 5, c = idx & 31;
            As[r][c] = cvt_bf16(A[(size_t)(m0 + r) * K + k0 + c]);
        }
        #pragma unroll
        for (int e = 0; e < 16; ++e) {
            int idx = tid + 256 * e;
            int r = idx >> 7, c = idx & 127;
            Bs[r][c] = cvt_bf16(B[(size_t)(k0 + r) * N + n0 + c]);
        }
        __syncthreads();

        v16u af[2], bfv[4];
        #pragma unroll
        for (int i = 0; i < 2; ++i) {
            int rr = wm * 32 + i * 16 + arow;
            #pragma unroll
            for (int j = 0; j < 8; ++j) {
                af[i][j]     = As[rr][kh + j];
                af[i][8 + j] = As[rr][16 + kh + j];
            }
        }
        #pragma unroll
        for (int j = 0; j < 4; ++j) {
            int cc = wn * 64 + j * 16 + bcol;
            #pragma unroll
            for (int e = 0; e < 16; ++e)
                bfv[j][e] = Bs[kb + (e & 7) + ((e >> 3) << 4)][cc];
        }
        #pragma unroll
        for (int i = 0; i < 2; ++i)
            #pragma unroll
            for (int j = 0; j < 4; ++j)
                acc[i][j] = wmma_bf16(af[i], bfv[j], acc[i][j]);
        __syncthreads();
    }

    const int rh = (l >> 4) << 3;
    #pragma unroll
    for (int i = 0; i < 2; ++i)
        #pragma unroll
        for (int j = 0; j < 4; ++j)
            #pragma unroll
            for (int v = 0; v < 8; ++v) {
                int r = m0 + wm * 32 + i * 16 + v + rh;
                int c = n0 + wn * 64 + j * 16 + (l & 15);
                C[(size_t)r * N + c] = acc[i][j][v];
            }
}

// ---------------------------------------------------------------------------
// RMSNorm + RoPE. Q rows -> bf16 (b,h,t,d). K rows -> f32 k_new slice AND
// bf16 window buffer Kw (b,hk,t,d). One 128-thread block per 256-elem row.
// ---------------------------------------------------------------------------
__global__ __launch_bounds__(128) void qk_norm_rope(
    const float* __restrict__ Qf, const float* __restrict__ Kf,
    const float* __restrict__ qw, const float* __restrict__ kw,
    unsigned short* __restrict__ Qb, float* __restrict__ Knew,
    unsigned short* __restrict__ Kw,
    const int* __restrict__ pos_p, const int* __restrict__ ci_p) {
    __shared__ float red[4];
    const int row = blockIdx.x;
    const int i   = threadIdx.x;
    const bool isq = row < (B_ * T_ * NH);

    const float* src; const float* w; int b, t, h;
    if (isq) {
        int r = row; h = r & (NH - 1); int bt = r >> 3;
        t = bt & (T_ - 1); b = bt >> 10;
        src = Qf + (size_t)bt * (NH * HD) + h * HD;  w = qw;
    } else {
        int r = row - B_ * T_ * NH; h = r & (NKV - 1); int bt = r >> 2;
        t = bt & (T_ - 1); b = bt >> 10;
        src = Kf + (size_t)bt * (NKV * HD) + h * HD; w = kw;
    }

    float x1 = src[i], x2 = src[i + 128];
    float ss = x1 * x1 + x2 * x2;
    #pragma unroll
    for (int m = 1; m < 32; m <<= 1) ss += __shfl_xor(ss, m, 32);
    if ((i & 31) == 0) red[i >> 5] = ss;
    __syncthreads();
    float tot = red[0] + red[1] + red[2] + red[3];
    float rms = rsqrtf(tot * (1.0f / 256.0f) + 1e-6f);

    float xn1 = x1 * rms * (1.0f + w[i]);
    float xn2 = x2 * rms * (1.0f + w[i + 128]);

    float pos      = (float)(pos_p[0] + t);
    float inv_freq = __expf(-((float)i * (1.0f / 128.0f)) * 9.210340372f);
    float ang = pos * inv_freq;
    float s, c; __sincosf(ang, &s, &c);
    float o1 = xn1 * c - xn2 * s;
    float o2 = xn2 * c + xn1 * s;

    if (isq) {
        unsigned short* dst = Qb + (((size_t)(b * NH + h)) * T_ + t) * HD;
        dst[i]       = f32_to_bf16(o1);
        dst[i + 128] = f32_to_bf16(o2);
    } else {
        float* dst = Knew + (((size_t)(b * NKV + h)) * SCACHE + ci_p[0] + t) * HD;
        dst[i]       = o1;
        dst[i + 128] = o2;
        unsigned short* db = Kw + (((size_t)(b * NKV + h)) * T_ + t) * HD;
        db[i]       = f32_to_bf16(o1);
        db[i + 128] = f32_to_bf16(o2);
    }
}

// Scatter V into v_new f32 slice AND transposed bf16 window VwT (b,hk,d,t).
__global__ __launch_bounds__(256) void v_update(
    const float* __restrict__ Vf, float* __restrict__ Vnew,
    unsigned short* __restrict__ VwT, const int* __restrict__ ci_p) {
    size_t idx = (size_t)blockIdx.x * 256 + threadIdx.x;
    int d  = idx & (HD - 1);
    int h  = (idx >> 8) & (NKV - 1);
    int bt = (int)(idx >> 10);
    int t  = bt & (T_ - 1);
    int b  = bt >> 10;
    float val = Vf[idx];
    Vnew[(((size_t)(b * NKV + h)) * SCACHE + ci_p[0] + t) * HD + d] = val;
    VwT[(((size_t)(b * NKV + h)) * HD + d) * T_ + t] = f32_to_bf16(val);
}

// ---------------------------------------------------------------------------
// Fused flash attention over the 1024-key window.
// Block = 128 threads (4 waves) per 16 query rows of one q-head; each wave
// owns 64 output head-dims. All staging via TDM (tensor_load_to_lds) with
// LDS pad-swizzle; TENSORcnt + workgroup barrier for sync.
// LDS tiles: Ks rows 512B + 8B pad -> [32][260]; VsT rows 64B + 8B -> [256][36];
// Qs like Ks -> [16][260].
// ---------------------------------------------------------------------------
__global__ __launch_bounds__(128) void flash_attn(
    const unsigned short* __restrict__ Qb,   // (B,NH,T,HD)   bf16
    const unsigned short* __restrict__ Kw,   // (B,NKV,T,HD)  bf16
    const unsigned short* __restrict__ VwT,  // (B,NKV,HD,T)  bf16 transposed
    unsigned short* __restrict__ Ob) {       // (B*T, NH*HD)  bf16
    __shared__ unsigned short Qs[16][260];
    __shared__ unsigned short Ks[32][260];
    __shared__ unsigned short VsT[256][36];
    __shared__ unsigned short Ps[4][16][32];

    const int tid  = threadIdx.x;
    const int l    = tid & 31;
    const int wave = tid >> 5;

    const int bid = blockIdx.x;               // B*NH*(T/16) = 1024 blocks
    const int t0  = (bid & 63) << 4;
    const int hq  = (bid >> 6) & (NH - 1);
    const int b   = bid >> 9;
    const int hk  = hq >> 1;                  // G = NH/NKV = 2

    const unsigned ldsQ = (unsigned)(uintptr_t)&Qs[0][0];
    const unsigned ldsK = (unsigned)(uintptr_t)&Ks[0][0];
    const unsigned ldsV = (unsigned)(uintptr_t)&VsT[0][0];

    const unsigned short* qsrc = Qb  + (((size_t)(b * NH  + hq)) * T_) * HD + (size_t)t0 * HD;
    const unsigned short* kbase = Kw  + (((size_t)(b * NKV + hk)) * T_) * HD;
    const unsigned short* vbase = VwT + (((size_t)(b * NKV + hk)) * HD) * T_;

    // Stage Q tile (16x256): pad 8B per 512B row -> matches [16][260].
    if (wave == 0) {
        tdm_load_2d_bf16(ldsQ, qsrc, /*tile*/256, 16, /*tensor*/256, 16, 256,
                         /*pad_int(512B)*/6, /*pad_amt(8B)*/1);
        __builtin_amdgcn_s_wait_tensorcnt(0);
    }
    __syncthreads();

    const int arow = l & 15;
    const int kh   = (l >> 4) << 3;
    const int kb   = (l >> 4) << 3;
    const int bcol = l & 15;

    v16u qf[8];
    #pragma unroll
    for (int kc = 0; kc < 8; ++kc)
        #pragma unroll
        for (int j = 0; j < 8; ++j) {
            qf[kc][j]     = Qs[arow][kc * 32 + kh + j];
            qf[kc][8 + j] = Qs[arow][kc * 32 + 16 + kh + j];
        }

    v8f   oacc[4] = {};
    float mrow[8], lrow[8];
    #pragma unroll
    for (int v = 0; v < 8; ++v) { mrow[v] = -3.0e38f; lrow[v] = 0.0f; }

    for (int ch = 0; ch < 32; ++ch) {         // 32 chunks x 32 keys = 1024
        if (wave == 0) {
            // K chunk: 32 rows x 256 cols from (t=ch*32 .. +32)
            tdm_load_2d_bf16(ldsK, kbase + (size_t)ch * 32 * HD,
                             256, 32, 256, 32, 256, 6, 1);
            // V chunk transposed: 256 rows x 32 keys, row length T_=1024
            tdm_load_2d_bf16(ldsV, vbase + (size_t)ch * 32,
                             32, 256, 1024, 256, 1024, /*pad_int(64B)*/3, 1);
            __builtin_amdgcn_s_wait_tensorcnt(0);
        }
        __syncthreads();

        // Scores: S(16x32) = Q(16x256) * K^T
        v8f s0 = {}, s1 = {};
        #pragma unroll
        for (int kc = 0; kc < 8; ++kc) {
            v16u b0, b1;
            #pragma unroll
            for (int e = 0; e < 16; ++e) {
                b0[e] = Ks[bcol][kc * 32 + kb + (e & 7) + ((e >> 3) << 4)];
                b1[e] = Ks[16 + bcol][kc * 32 + kb + (e & 7) + ((e >> 3) << 4)];
            }
            s0 = wmma_bf16(qf[kc], b0, s0);
            s1 = wmma_bf16(qf[kc], b1, s1);
        }

        // Online softmax (C layout: row = v + 8*(l>>4), col = l&15).
        float x[8], corr[8];
        #pragma unroll
        for (int v = 0; v < 8; ++v) {
            float a0 = s0[v] * SCALE, a1 = s1[v] * SCALE;
            s0[v] = a0; s1[v] = a1;
            x[v] = fmaxf(a0, a1);
        }
        #pragma unroll
        for (int m = 1; m <= 8; m <<= 1)
            #pragma unroll
            for (int v = 0; v < 8; ++v)
                x[v] = fmaxf(x[v], __shfl_xor(x[v], m, 32));
        #pragma unroll
        for (int v = 0; v < 8; ++v) {
            float mn = fmaxf(mrow[v], x[v]);
            corr[v] = __expf(mrow[v] - mn);
            mrow[v] = mn;
            s0[v] = __expf(s0[v] - mn);
            s1[v] = __expf(s1[v] - mn);
            x[v]  = s0[v] + s1[v];
        }
        #pragma unroll
        for (int m = 1; m <= 8; m <<= 1)
            #pragma unroll
            for (int v = 0; v < 8; ++v)
                x[v] += __shfl_xor(x[v], m, 32);
        #pragma unroll
        for (int v = 0; v < 8; ++v) lrow[v] = lrow[v] * corr[v] + x[v];
        #pragma unroll
        for (int j = 0; j < 4; ++j)
            #pragma unroll
            for (int v = 0; v < 8; ++v) oacc[j][v] *= corr[v];

        // Relayout P: C layout -> A layout via per-wave LDS tile
        // (same-wave LDS ops are in-order; no barrier needed).
        const int ph = (l >> 4) << 3;
        #pragma unroll
        for (int v = 0; v < 8; ++v) {
            Ps[wave][v + ph][l & 15]        = f32_to_bf16(s0[v]);
            Ps[wave][v + ph][16 + (l & 15)] = f32_to_bf16(s1[v]);
        }
        v16u pa;
        #pragma unroll
        for (int j = 0; j < 8; ++j) {
            pa[j]     = Ps[wave][arow][kh + j];
            pa[8 + j] = Ps[wave][arow][16 + kh + j];
        }

        // O(16 x 64-per-wave) += P(16x32) * V(32x64); V frag reads contiguous.
        const int d0 = wave * 64;
        #pragma unroll
        for (int j = 0; j < 4; ++j) {
            v16u vb;
            #pragma unroll
            for (int e = 0; e < 16; ++e)
                vb[e] = VsT[d0 + j * 16 + bcol][kb + (e & 7) + ((e >> 3) << 4)];
            oacc[j] = wmma_bf16(pa, vb, oacc[j]);
        }
        __syncthreads();
    }

    float inv[8];
    #pragma unroll
    for (int v = 0; v < 8; ++v) inv[v] = 1.0f / lrow[v];
    #pragma unroll
    for (int j = 0; j < 4; ++j)
        #pragma unroll
        for (int v = 0; v < 8; ++v) {
            int t = t0 + v + ((l >> 4) << 3);
            int c = hq * HD + wave * 64 + j * 16 + (l & 15);
            Ob[((size_t)(b * T_ + t)) * (NH * HD) + c] =
                f32_to_bf16(oacc[j][v] * inv[v]);
        }
}

// ---------------------------------------------------------------------------
extern "C" void kernel_launch(void* const* d_in, const int* in_sizes, int n_in,
                              void* d_out, int out_size, void* d_ws, size_t ws_size,
                              hipStream_t stream) {
    (void)in_sizes; (void)n_in; (void)out_size; (void)ws_size;

    const float* x       = (const float*)d_in[0];
    const float* k_cache = (const float*)d_in[1];
    const float* v_cache = (const float*)d_in[2];
    const float* wq      = (const float*)d_in[3];
    const float* wk      = (const float*)d_in[4];
    const float* wv      = (const float*)d_in[5];
    const float* wo      = (const float*)d_in[6];
    const float* qnw     = (const float*)d_in[7];
    const float* knw     = (const float*)d_in[8];
    const int*   pos_p   = (const int*)d_in[9];
    const int*   ci_p    = (const int*)d_in[10];

    float* out  = (float*)d_out;                       // (B,T,HID)
    float* knew = out  + (size_t)B_ * T_ * HID;        // (B,NKV,SCACHE,HD)
    float* vnew = knew + (size_t)B_ * NKV * SCACHE * HD;

    // Workspace carve-up (56 MB total)
    float* Qf = (float*)d_ws;                          // (B*T, NH*HD)  f32
    float* Kf = Qf + (size_t)B_ * T_ * NH * HD;        // (B*T, NKV*HD) f32
    float* Vf = Kf + (size_t)B_ * T_ * NKV * HD;       // (B*T, NKV*HD) f32
    unsigned short* Qb  = (unsigned short*)(Vf + (size_t)B_ * T_ * NKV * HD);
    unsigned short* Ab  = Qb  + (size_t)B_ * NH  * T_ * HD;  // attn out bf16
    unsigned short* Kw  = Ab  + (size_t)B_ * T_ * NH * HD;   // K window bf16
    unsigned short* VwT = Kw  + (size_t)B_ * NKV * T_ * HD;  // V window bf16 (T-major)

    const size_t cacheBytes = (size_t)B_ * NKV * SCACHE * HD * sizeof(float);
    hipMemcpyAsync(knew, k_cache, cacheBytes, hipMemcpyDeviceToDevice, stream);
    hipMemcpyAsync(vnew, v_cache, cacheBytes, hipMemcpyDeviceToDevice, stream);

    // QKV projections: (2048 x 2560) * (2560 x {2048,1024,1024})
    gemm_wmma<float, float><<<dim3(16, 16), 256, 0, stream>>>(x, wq, Qf, 2048, 2048, 2560);
    gemm_wmma<float, float><<<dim3(8, 16), 256, 0, stream>>>(x, wk, Kf, 2048, 1024, 2560);
    gemm_wmma<float, float><<<dim3(8, 16), 256, 0, stream>>>(x, wv, Vf, 2048, 1024, 2560);

    // RMSNorm + RoPE (Q -> bf16; K -> f32 slice + bf16 window), V scatter
    qk_norm_rope<<<B_ * T_ * (NH + NKV), 128, 0, stream>>>(
        Qf, Kf, qnw, knw, Qb, knew, Kw, pos_p, ci_p);
    v_update<<<(B_ * T_ * NKV * HD) / 256, 256, 0, stream>>>(Vf, vnew, VwT, ci_p);

    // Fused attention over the 1024-key window (TDM-staged)
    flash_attn<<<B_ * NH * (T_ / 16), 128, 0, stream>>>(Qb, Kw, VwT, Ab);

    // Output projection: (2048 x 2048 bf16) * (2048 x 2560 f32) -> d_out
    gemm_wmma<unsigned short, float><<<dim3(20, 16), 256, 0, stream>>>(Ab, wo, out, 2048, 2560, 2048);
}
